// DiseaseGNN_68444598829350
// MI455X (gfx1250) — matrix operand order repaired and checked by reading, compiled, not run on hardware
//
#include <hip/hip_runtime.h>
#include <hip/hip_bf16.h>

// ---------------------------------------------------------------------------
// DiseaseGNN on MI455X (gfx1250, wave32, WMMA).
//
// Pipeline per GCN layer:
//   hs  = (X @ W) * dinv         (WMMA f32_16x16x32_f16 GEMM, K=64)
//   acc = hs                      (self-loop term)
//   acc[dst] += hs[src]           (edge scatter, float4 gather + f32 atomics)
//   X'  = relu(dinv * acc + b)    (finalize)
// Head: relu(h @ Wm1 + bm1) via WMMA, then sigmoid(dot(h, Wm2)+bm2) via
// wave32 shuffle reduction.
// ---------------------------------------------------------------------------

typedef __attribute__((ext_vector_type(16))) _Float16 v16h;
typedef __attribute__((ext_vector_type(8)))  float    v8f;

#define HID 64

// ---------------- degree / normalization ----------------

__global__ void k_deg_init(unsigned int* degcnt, int N) {
    int i = blockIdx.x * blockDim.x + threadIdx.x;
    if (i < N) degcnt[i] = 1u;  // self loop
}

__global__ void k_deg_count(const int* __restrict__ ei, long long E,
                            unsigned int* degcnt) {
    long long e = (long long)blockIdx.x * blockDim.x + threadIdx.x;
    if (e < E) atomicAdd(&degcnt[ei[E + e]], 1u);  // dst side
}

__global__ void k_dinv(const unsigned int* __restrict__ degcnt,
                       float* __restrict__ dinv, int N) {
    int i = blockIdx.x * blockDim.x + threadIdx.x;
    if (i < N) dinv[i] = rsqrtf((float)degcnt[i]);
}

// ---------------- weight prep: f32 (K,64) -> f16 transposed (64,K) ----------

__global__ void k_w_to_half_T(const float* __restrict__ W, _Float16* __restrict__ WhT) {
    int t = blockIdx.x * blockDim.x + threadIdx.x;  // t < 4096
    int k = t >> 6, n = t & 63;
    WhT[n * 64 + k] = (_Float16)W[t];
}

// ---------------- layer 1: K=3 scalar GEMM + dinv scale, dual write --------

__global__ void k_layer1(const float* __restrict__ x, const float* __restrict__ W1,
                         const float* __restrict__ dinv,
                         float* __restrict__ hs, float* __restrict__ acc, int N) {
    long long t = (long long)blockIdx.x * blockDim.x + threadIdx.x;
    int i = (int)(t >> 6), c = (int)(t & 63);
    if (i >= N) return;
    float h = x[i * 3 + 0] * W1[c] + x[i * 3 + 1] * W1[64 + c] + x[i * 3 + 2] * W1[128 + c];
    float v = h * dinv[i];
    size_t off = (size_t)i * HID + c;
    hs[off]  = v;
    acc[off] = v;
}

// ---------------- WMMA GEMM: out = X(N,64) @ W(64,64) --------------------
// MODE 0: GCN layer -> v = gemm * dinv[row]; write hs and acc (self loop)
// MODE 1: MLP layer -> v = relu(gemm + bias[col]); write out0 only

template <int MODE>
__global__ __launch_bounds__(256) void k_gemm64(
    const float* __restrict__ X, const _Float16* __restrict__ WhT,
    const float* __restrict__ dinv, const float* __restrict__ bias,
    float* __restrict__ out0, float* __restrict__ out1, int N) {

    __shared__ _Float16 lw[64 * 64];  // W transposed: lw[n*64 + k], 8 KB

    // cooperative coalesced LDS stage of the (tiny) weight matrix
    {
        const unsigned int* src = (const unsigned int*)WhT;
        unsigned int* dst = (unsigned int*)lw;
        for (int t = threadIdx.x; t < 2048; t += 256) dst[t] = src[t];
    }
    __syncthreads();

    const int wave  = threadIdx.x >> 5;
    const int lane  = threadIdx.x & 31;
    const int lanem = lane & 15;
    const int hiK   = lane >> 4;      // lane group 0/1
    const int kbA   = hiK * 16;       // A-fragment K base (contiguous, matches B)
    const int kbB   = hiK * 16;       // B-fragment K base (contiguous, matches A)

    const int m0  = blockIdx.x * 128 + wave * 16;  // 8 waves x 16 rows
    int row = m0 + lanem;
    int rowc = row < N ? row : 0;  // clamp loads; EXEC stays all-ones for WMMA

    v8f acc[4] = {};

    #pragma unroll
    for (int kt = 0; kt < 64; kt += 32) {
        // A fragment: idx 0..15 -> K = kt+kbA+idx
        const float* xr = X + (size_t)rowc * HID + kt + kbA;
        float4 p0 = *(const float4*)(xr + 0);
        float4 p1 = *(const float4*)(xr + 4);
        float4 p2 = *(const float4*)(xr + 8);
        float4 p3 = *(const float4*)(xr + 12);
        v16h a;
        a[0]  = (_Float16)p0.x; a[1]  = (_Float16)p0.y; a[2]  = (_Float16)p0.z; a[3]  = (_Float16)p0.w;
        a[4]  = (_Float16)p1.x; a[5]  = (_Float16)p1.y; a[6]  = (_Float16)p1.z; a[7]  = (_Float16)p1.w;
        a[8]  = (_Float16)p2.x; a[9]  = (_Float16)p2.y; a[10] = (_Float16)p2.z; a[11] = (_Float16)p2.w;
        a[12] = (_Float16)p3.x; a[13] = (_Float16)p3.y; a[14] = (_Float16)p3.z; a[15] = (_Float16)p3.w;

        #pragma unroll
        for (int nt = 0; nt < 4; nt++) {
            // B fragment: lane n = nt*16+lanem, K = kt+kbB .. +15 contiguous in LDS
            const v16h b = *(const v16h*)(&lw[(nt * 16 + lanem) * 64 + kt + kbB]);
            acc[nt] = __builtin_amdgcn_wmma_f32_16x16x32_f16(
                false, a, false, b, (short)0, acc[nt], false, false);
        }
    }

    // C/D layout: vgpr r, lanes 0-15 -> M=r, lanes 16-31 -> M=r+8; N = lanem
    const int mh = hiK * 8;
    #pragma unroll
    for (int r = 0; r < 8; r++) {
        int orow = m0 + mh + r;
        if (orow >= N) continue;
        if (MODE == 0) {
            float dv = dinv[orow];
            #pragma unroll
            for (int nt = 0; nt < 4; nt++) {
                float v = acc[nt][r] * dv;
                size_t off = (size_t)orow * HID + nt * 16 + lanem;
                out0[off] = v;
                out1[off] = v;
            }
        } else {
            #pragma unroll
            for (int nt = 0; nt < 4; nt++) {
                int col = nt * 16 + lanem;
                float v = acc[nt][r] + bias[col];
                out0[(size_t)orow * HID + col] = v > 0.0f ? v : 0.0f;
            }
        }
    }
}

// ---------------- edge scatter: acc[dst] += hs[src], 16 lanes/edge ---------

__global__ void k_scatter(const int* __restrict__ ei, long long E,
                          const float* __restrict__ hs, float* __restrict__ acc) {
    unsigned long long t = (unsigned long long)blockIdx.x * blockDim.x + threadIdx.x;
    long long e = (long long)(t >> 4);
    if (e >= E) return;
    int c0 = (int)(t & 15) * 4;
    int s = ei[e];
    int d = ei[E + e];
    const float4 v = *(const float4*)(hs + (size_t)s * HID + c0);
    float* a = acc + (size_t)d * HID + c0;
    atomicAdd(a + 0, v.x);
    atomicAdd(a + 1, v.y);
    atomicAdd(a + 2, v.z);
    atomicAdd(a + 3, v.w);
}

// ---------------- finalize: X' = relu(dinv*acc + b) ------------------------

__global__ void k_finalize(const float* __restrict__ acc, const float* __restrict__ dinv,
                           const float* __restrict__ b, float* __restrict__ out, int N) {
    long long t = (long long)blockIdx.x * blockDim.x + threadIdx.x;
    int i = (int)(t >> 6), c = (int)(t & 63);
    if (i >= N) return;
    float v = dinv[i] * acc[t] + b[c];
    out[t] = v > 0.0f ? v : 0.0f;
}

// ---------------- head: sigmoid(dot(h, Wm2) + bm2), wave per node ----------

__global__ void k_head(const float* __restrict__ h, const float* __restrict__ wm2,
                       const float* __restrict__ bm2, float* __restrict__ out, int N) {
    int node = (blockIdx.x * blockDim.x + threadIdx.x) >> 5;
    int lane = threadIdx.x & 31;
    if (node >= N) return;
    float2 v = *(const float2*)(h + (size_t)node * HID + lane * 2);
    float2 w = *(const float2*)(wm2 + lane * 2);
    float p = v.x * w.x + v.y * w.y;
    #pragma unroll
    for (int o = 16; o > 0; o >>= 1) p += __shfl_xor(p, o, 32);
    if (lane == 0) out[node] = 1.0f / (1.0f + __expf(-(p + bm2[0])));
}

// ---------------------------------------------------------------------------

extern "C" void kernel_launch(void* const* d_in, const int* in_sizes, int n_in,
                              void* d_out, int out_size, void* d_ws, size_t ws_size,
                              hipStream_t stream) {
    const float* x   = (const float*)d_in[0];
    const int*   ei  = (const int*)d_in[1];
    const float* W1  = (const float*)d_in[2];
    const float* b1  = (const float*)d_in[3];
    const float* W2  = (const float*)d_in[4];
    const float* b2  = (const float*)d_in[5];
    const float* W3  = (const float*)d_in[6];
    const float* b3  = (const float*)d_in[7];
    const float* Wm1 = (const float*)d_in[8];
    const float* bm1 = (const float*)d_in[9];
    const float* Wm2 = (const float*)d_in[10];
    const float* bm2 = (const float*)d_in[11];

    const int       N = in_sizes[0] / 3;
    const long long E = in_sizes[1] / 2;

    // workspace layout (256B aligned slices)
    char* p = (char*)d_ws;
    auto alloc = [&](size_t bytes) {
        void* r = (void*)p;
        p += (bytes + 255) & ~(size_t)255;
        return r;
    };
    unsigned int* degcnt = (unsigned int*)alloc((size_t)N * 4);
    float*    dinv  = (float*)alloc((size_t)N * 4);
    float*    hs    = (float*)alloc((size_t)N * HID * 4);
    float*    accb  = (float*)alloc((size_t)N * HID * 4);
    float*    hA    = (float*)alloc((size_t)N * HID * 4);
    float*    hB    = (float*)alloc((size_t)N * HID * 4);
    _Float16* WhT2  = (_Float16*)alloc(64 * 64 * 2);
    _Float16* WhT3  = (_Float16*)alloc(64 * 64 * 2);
    _Float16* WhTm1 = (_Float16*)alloc(64 * 64 * 2);

    const int B = 256;
    const unsigned gN    = (unsigned)((N + B - 1) / B);
    const unsigned gE    = (unsigned)((E + B - 1) / B);
    const unsigned gNC   = (unsigned)(((long long)N * HID + B - 1) / B);
    const unsigned gScat = (unsigned)((E * 16 + B - 1) / B);
    const unsigned gGemm = (unsigned)((N + 127) / 128);
    const unsigned gHead = (unsigned)(((long long)N * 32 + B - 1) / B);

    // degrees -> dinv
    k_deg_init <<<gN, B, 0, stream>>>(degcnt, N);
    k_deg_count<<<gE, B, 0, stream>>>(ei, E, degcnt);
    k_dinv     <<<gN, B, 0, stream>>>(degcnt, dinv, N);

    // weight conversion (f16, transposed) for the three 64x64 GEMMs
    k_w_to_half_T<<<16, B, 0, stream>>>(W2, WhT2);
    k_w_to_half_T<<<16, B, 0, stream>>>(W3, WhT3);
    k_w_to_half_T<<<16, B, 0, stream>>>(Wm1, WhTm1);

    // ---- layer 1 (K=3, scalar GEMM) ----
    k_layer1  <<<gNC, B, 0, stream>>>(x, W1, dinv, hs, accb, N);
    k_scatter <<<gScat, B, 0, stream>>>(ei, E, hs, accb);
    k_finalize<<<gNC, B, 0, stream>>>(accb, dinv, b1, hA, N);

    // ---- layer 2 (WMMA GEMM) ----
    k_gemm64<0><<<gGemm, B, 0, stream>>>(hA, WhT2, dinv, nullptr, hs, accb, N);
    k_scatter  <<<gScat, B, 0, stream>>>(ei, E, hs, accb);
    k_finalize <<<gNC, B, 0, stream>>>(accb, dinv, b2, hB, N);

    // ---- layer 3 (WMMA GEMM) ----
    k_gemm64<0><<<gGemm, B, 0, stream>>>(hB, WhT3, dinv, nullptr, hs, accb, N);
    k_scatter  <<<gScat, B, 0, stream>>>(ei, E, hs, accb);
    k_finalize <<<gNC, B, 0, stream>>>(accb, dinv, b3, hA, N);

    // ---- MLP head ----
    k_gemm64<1><<<gGemm, B, 0, stream>>>(hA, WhTm1, nullptr, bm1, hB, nullptr, N);
    k_head     <<<gHead, B, 0, stream>>>(hB, Wm2, bm2, (float*)d_out, N);
}
